// Conv2D_41412074668081
// MI455X (gfx1250) — matrix operand order repaired and checked by reading, not compiled, not run on hardware
//
#include <hip/hip_runtime.h>

// ---- CDNA5 vector types (trivial, union-safe) ----
typedef __attribute__((ext_vector_type(16))) _Float16 v16h;
typedef __attribute__((ext_vector_type(8)))  float    v8f;
typedef __attribute__((ext_vector_type(4)))  int      i128;

#define C_IN   128
#define H_IMG  56
#define W_IMG  56
#define S_IMG  (H_IMG * W_IMG)      // 3136
#define OC_TOT 256
#define KDIM   (C_IN * 9)           // 1152
#define NPIX   (32 * S_IMG)         // 100352

#define M_TILE 128
#define N_TILE 128
#define KC     32                   // channels per K step
#define NSTEP  36                   // 9 positions * 4 channel chunks
#define APITCH 40                   // halves per LDS row (32 data + 8 pad)
#define BPITCH 40

union Frag { v16h v; struct { i128 lo, hi; } q; };

// Fetch one K-step (A: weights, B: im2col input) into registers.
// All loads are unconditional and in-bounds; invalid halo pixels are
// neutralized with `scale` at conversion time.
__device__ __forceinline__ void fetch_step(int s,
    const float* __restrict__ xN, const float* __restrict__ wBase,
    int ohB, int owB, int cgB,
    float vA[16], float vB[16], float& scale)
{
    const int pos = s >> 2;
    const int c0  = (s & 3) * KC;
    const int kh  = pos / 3;
    const int kw  = pos - kh * 3;
    const int ih  = ohB + kh - 1;
    const int iw  = owB + kw - 1;
    const bool valid = ((unsigned)ih < (unsigned)H_IMG) &&
                       ((unsigned)iw < (unsigned)W_IMG);
    scale = valid ? 1.0f : 0.0f;
    const int soff = valid ? (ih * W_IMG + iw) : 0;   // clamp -> in-bounds

    const float* pB = xN + (size_t)(c0 + cgB * 16) * S_IMG + soff;
    #pragma unroll
    for (int i = 0; i < 16; ++i) vB[i] = pB[(size_t)i * S_IMG];

    const float* pA = wBase + pos + c0 * 9;
    #pragma unroll
    for (int i = 0; i < 16; ++i) vA[i] = pA[(size_t)i * KDIM];
}

// Convert to f16 and store into the given LDS buffers.
__device__ __forceinline__ void stage_store(
    _Float16* __restrict__ As, _Float16* __restrict__ Bs,
    int pB_, int cgB, int mgA, int kcA,
    const float vA[16], const float vB[16], float scale)
{
    union { _Float16 h[16]; i128 q[2]; } tb;
    #pragma unroll
    for (int i = 0; i < 16; ++i) tb.h[i] = (_Float16)(vB[i] * scale);
    i128* dst = (i128*)&Bs[pB_ * BPITCH + cgB * 16];
    dst[0] = tb.q[0];
    dst[1] = tb.q[1];

    #pragma unroll
    for (int i = 0; i < 16; ++i)
        As[(mgA * 16 + i) * APITCH + kcA] = (_Float16)vA[i];
}

__global__ __launch_bounds__(256)
void Conv2D_wmma_f16(const float* __restrict__ x,
                     const float* __restrict__ Wm,
                     const float* __restrict__ bias,
                     float* __restrict__ out)
{
    __shared__ __align__(16) _Float16 AsBuf[2][M_TILE * APITCH]; // [oc][kc]
    __shared__ __align__(16) _Float16 BsBuf[2][N_TILE * BPITCH]; // [pix][kc]
    __shared__ float biasS[M_TILE];

    const int t    = threadIdx.x;
    const int oc0  = blockIdx.y * M_TILE;
    const int pix0 = blockIdx.x * N_TILE;

    if (t < M_TILE) biasS[t] = bias[oc0 + t];

    // ---- B-tile loader mapping: 128 pixels x 32 channels, 16 elems/thread
    const int pB  = t & 127;        // pixel within tile
    const int cgB = t >> 7;         // channel half-group (0..1)
    const int pxB = pix0 + pB;
    const int nB  = pxB / S_IMG;
    const int sB  = pxB - nB * S_IMG;
    const int ohB = sB / W_IMG;
    const int owB = sB - ohB * W_IMG;
    const float* xN = x + (size_t)nB * C_IN * S_IMG;

    // ---- A-tile loader mapping: 128 oc x 32 kc, 16 elems/thread
    const int kcA = t & 31;
    const int mgA = t >> 5;         // 0..7
    const float* wBase = Wm + (size_t)(oc0 + mgA * 16) * KDIM + kcA * 9;

    // ---- wave / lane mapping (wave32)
    const int lane   = t & 31;
    const int wid    = t >> 5;
    const int waveM  = wid & 1;     // 2 waves over M (64 rows each)
    const int waveN  = wid >> 1;    // 4 waves over N (32 cols each)
    const int lane16 = lane & 15;
    const int halfId = lane >> 4;

    const v8f vzero = {0.f, 0.f, 0.f, 0.f, 0.f, 0.f, 0.f, 0.f};
    v8f acc[4][2];
    #pragma unroll
    for (int mt = 0; mt < 4; ++mt)
        #pragma unroll
        for (int nt = 0; nt < 2; ++nt) acc[mt][nt] = vzero;

    // ---- prologue: fetch + stage step 0 ----
    float vA[16], vB[16], sc;
    fetch_step(0, xN, wBase, ohB, owB, cgB, vA, vB, sc);
    stage_store(AsBuf[0], BsBuf[0], pB, cgB, mgA, kcA, vA, vB, sc);
    __syncthreads();

    int buf = 0;
    #pragma unroll 1
    for (int s = 0; s < NSTEP; ++s) {
        const bool more = (s + 1 < NSTEP);
        float nA[16], nB16[16], nsc;
        if (more)   // prefetch next step while this step computes
            fetch_step(s + 1, xN, wBase, ohB, owB, cgB, nA, nB16, nsc);

        const _Float16* Asb = AsBuf[buf];
        const _Float16* Bsb = BsBuf[buf];

        // ---- fetch operands in ISA 16-bit WMMA layout ----
        // A (16x32, MxK): lane=row M; lanes<16: K0-7 @+0, K16-23 @+32
        //                 lanes>=16: K8-15 @+16, K24-31 @+48
        Frag a[4], b[2];
        #pragma unroll
        for (int mt = 0; mt < 4; ++mt) {
            const char* base = (const char*)Asb
                + (waveM * 64 + mt * 16 + lane16) * (APITCH * 2)
                + halfId * 16;
            a[mt].q.lo = *(const i128*)(base);
            a[mt].q.hi = *(const i128*)(base + 32);
        }
        // B (32x16, KxN): lane=col N; lanes<16: K0-7,K16-23; lanes>=16: K8-15,K24-31
        #pragma unroll
        for (int nt = 0; nt < 2; ++nt) {
            const char* base = (const char*)Bsb
                + (waveN * 32 + nt * 16 + lane16) * (BPITCH * 2)
                + halfId * 16;
            b[nt].q.lo = *(const i128*)(base);
            b[nt].q.hi = *(const i128*)(base + 32);
        }

        #pragma unroll
        for (int mt = 0; mt < 4; ++mt)
            #pragma unroll
            for (int nt = 0; nt < 2; ++nt)
                acc[mt][nt] = __builtin_amdgcn_wmma_f32_16x16x32_f16(
                    false, a[mt].v, false, b[nt].v,
                    (short)0, acc[mt][nt], false, false);

        if (more) {
            stage_store(AsBuf[buf ^ 1], BsBuf[buf ^ 1],
                        pB, cgB, mgA, kcA, nA, nB16, nsc);
            __syncthreads();
            buf ^= 1;
        }
    }

    // ---- epilogue: D layout -> NCHW stores, bias from LDS ----
    #pragma unroll
    for (int nt = 0; nt < 2; ++nt) {
        const int px = pix0 + waveN * 32 + nt * 16 + lane16;
        const int n  = px / S_IMG;
        const int s  = px - n * S_IMG;
        float* o = out + ((size_t)n * OC_TOT + oc0) * S_IMG + s;
        #pragma unroll
        for (int mt = 0; mt < 4; ++mt) {
            const int ocRel = waveM * 64 + mt * 16 + halfId * 8;
            #pragma unroll
            for (int r = 0; r < 8; ++r)
                o[(size_t)(ocRel + r) * S_IMG] =
                    acc[mt][nt][r] + biasS[ocRel + r];
        }
    }
}

extern "C" void kernel_launch(void* const* d_in, const int* in_sizes, int n_in,
                              void* d_out, int out_size, void* d_ws, size_t ws_size,
                              hipStream_t stream) {
    const float* x  = (const float*)d_in[0];
    const float* Wm = (const float*)d_in[1];
    const float* b  = (const float*)d_in[2];
    float* out = (float*)d_out;

    dim3 grid(NPIX / N_TILE, OC_TOT / M_TILE);   // (784, 2)
    Conv2D_wmma_f16<<<grid, 256, 0, stream>>>(x, Wm, b, out);
}
